// YoloV3_54400055771390
// MI455X (gfx1250) — matrix operand order repaired and checked by reading, not compiled, not run on hardware
//
#include <hip/hip_runtime.h>

typedef __attribute__((ext_vector_type(16))) __bf16 v16bf;
typedef __attribute__((ext_vector_type(8)))  __bf16 v8bf;
typedef __attribute__((ext_vector_type(8)))  float  v8f;
typedef __attribute__((ext_vector_type(4)))  int    v4i;
typedef __attribute__((address_space(1))) v4i* gv4i_p;   // global (AS1) int4*
typedef __attribute__((address_space(3))) v4i* lv4i_p;   // LDS (AS3) int4*

#define GN_EPS 1e-3f

#if defined(__has_builtin)
#  if __has_builtin(__builtin_amdgcn_global_load_async_to_lds_b128) && \
      __has_builtin(__builtin_amdgcn_s_wait_asynccnt)
#    define CDNA5_ASYNC 1
#  endif
#endif
#ifndef CDNA5_ASYNC
#  define CDNA5_ASYNC 0
#endif

// ---------------------------------------------------------------------------
// Weight repack: wgt (k,k,Cin,Cout) f32 -> wrep[n][k] bf16, K padded to KtotPad
// ---------------------------------------------------------------------------
__global__ void repack_w(const float* __restrict__ wgt, __bf16* __restrict__ wrep,
                         int Ktot, int KtotPad, int Cout)
{
  long N = (long)Cout * KtotPad;
  long i = (long)blockIdx.x * blockDim.x + threadIdx.x;
  if (i >= N) return;
  int n = (int)(i / KtotPad);
  int k = (int)(i % KtotPad);
  float v = (k < Ktot) ? wgt[(size_t)k * Cout + n] : 0.f;
  wrep[i] = (__bf16)v;
}

__global__ void f32_to_bf16(const float* __restrict__ x, __bf16* __restrict__ y, long N)
{
  long i = (long)blockIdx.x * blockDim.x + threadIdx.x;
  if (i < N) y[i] = (__bf16)x[i];
}

// ---- helpers -------------------------------------------------------------
__device__ __forceinline__ v16bf packFrag(v8bf lo, v8bf hi)
{
  v16bf r;
#pragma unroll
  for (int j = 0; j < 8; ++j) { r[j] = lo[j]; r[j + 8] = hi[j]; }
  return r;
}

#define LDSB_STRIDE 40   // bf16 elems per row (80B): conflict-free b128 reads

// Fill one 64(N) x 32(K) bf16 chunk of B into LDS (4KB, one b128 per thread).
__device__ __forceinline__ void fillB(const __bf16* __restrict__ wrep, __bf16* lds,
                                      int nBase, int Cout, int KtotPad, int kg, int tid)
{
  int n = tid >> 2, part = tid & 3;
  int row = nBase + n; if (row >= Cout) row = Cout - 1;
  const __bf16* gp = wrep + (size_t)row * KtotPad + kg + part * 8;
  __bf16* lp = lds + n * LDSB_STRIDE + part * 8;
#if CDNA5_ASYNC
  __builtin_amdgcn_global_load_async_to_lds_b128(
      (gv4i_p)(void*)gp, (lv4i_p)(void*)lp, 0, 0);
#else
  *(v8bf*)lp = *(const v8bf*)gp;
#endif
}

__device__ __forceinline__ void waitAsync0()
{
#if CDNA5_ASYNC
  __builtin_amdgcn_s_wait_asynccnt(0);
#endif
}
__device__ __forceinline__ void waitAsync1()
{
#if CDNA5_ASYNC
  __builtin_amdgcn_s_wait_asynccnt(1);
#endif
}

// ---------------------------------------------------------------------------
// Fast implicit-GEMM conv (Cin % 32 == 0): per-wave 16(M)x64(N) tile,
// 4 accumulators; B chunks staged in LDS (double-buffered, async-to-LDS),
// A loaded directly as contiguous b128 runs. bf16 in, f32 out.
// ---------------------------------------------------------------------------
__global__ __launch_bounds__(256) void conv_wmma_fast(
    const __bf16* __restrict__ in, const __bf16* __restrict__ wrep,
    const float* __restrict__ bias, float* __restrict__ out,
    int B, int H, int W, int Cin, int Cout, int K, int stride,
    int Hout, int Wout, int padLo, int KtotPad)
{
  __shared__ __bf16 ldsB[2][64 * LDSB_STRIDE];
  const int tid   = threadIdx.x;
  const int lane  = tid & 31;
  const int wave  = tid >> 5;
  const int tileM = blockIdx.x * 8 + wave;
  const int M     = B * Hout * Wout;

  const int mr = lane & 15;
  const int m  = tileM * 16 + mr;
  int bb = 0, oh = 0, ow = 0;
  if (m < M) { bb = m / (Hout * Wout); int r = m % (Hout * Wout); oh = r / Wout; ow = r % Wout; }
  const int aKoff = (lane < 16) ? 0 : 16;   // A fragment K base (contiguous k = 16h + j, same order as B)
  const int bKoff = (lane < 16) ? 0 : 16;   // B fragment K base
  const int nBase = blockIdx.y * 64;
  const int nq    = (K * K * Cin) >> 5;     // number of 32-K chunks

  fillB(wrep, ldsB[0], nBase, Cout, KtotPad, 0, tid);
  waitAsync0();
  __syncthreads();

  v8f acc0 = {}, acc1 = {}, acc2 = {}, acc3 = {};
  int q = 0;
  for (int kh = 0; kh < K; ++kh) {
    int ih = oh * stride + kh - padLo;
    for (int kw = 0; kw < K; ++kw) {
      int iw = ow * stride + kw - padLo;
      bool ok = (m < M) && ((unsigned)ih < (unsigned)H) && ((unsigned)iw < (unsigned)W);
      const __bf16* ap = in + (((size_t)bb * H + ih) * W + iw) * Cin;
      for (int c0 = 0; c0 < Cin; c0 += 32) {
        const int cur = q & 1;
        if (q + 1 < nq)                       // overlap next fill with compute
          fillB(wrep, ldsB[cur ^ 1], nBase, Cout, KtotPad, (q + 1) * 32, tid);

        v16bf av;
#pragma unroll
        for (int j = 0; j < 16; ++j) av[j] = (__bf16)0.0f;
        if (ok) {                             // two contiguous 16B runs -> b128
          v8bf lo = *(const v8bf*)(ap + c0 + aKoff);
          v8bf hi = *(const v8bf*)(ap + c0 + aKoff + 8);
          av = packFrag(lo, hi);
        }
        const __bf16* lb = ldsB[cur] + (lane & 15) * LDSB_STRIDE + bKoff;
        v16bf bv0 = packFrag(*(const v8bf*)(lb + 0 * 16 * LDSB_STRIDE),
                             *(const v8bf*)(lb + 0 * 16 * LDSB_STRIDE + 8));
        v16bf bv1 = packFrag(*(const v8bf*)(lb + 1 * 16 * LDSB_STRIDE),
                             *(const v8bf*)(lb + 1 * 16 * LDSB_STRIDE + 8));
        v16bf bv2 = packFrag(*(const v8bf*)(lb + 2 * 16 * LDSB_STRIDE),
                             *(const v8bf*)(lb + 2 * 16 * LDSB_STRIDE + 8));
        v16bf bv3 = packFrag(*(const v8bf*)(lb + 3 * 16 * LDSB_STRIDE),
                             *(const v8bf*)(lb + 3 * 16 * LDSB_STRIDE + 8));
        acc0 = __builtin_amdgcn_wmma_f32_16x16x32_bf16(false, av, false, bv0, (short)0, acc0, false, false);
        acc1 = __builtin_amdgcn_wmma_f32_16x16x32_bf16(false, av, false, bv1, (short)0, acc1, false, false);
        acc2 = __builtin_amdgcn_wmma_f32_16x16x32_bf16(false, av, false, bv2, (short)0, acc2, false, false);
        acc3 = __builtin_amdgcn_wmma_f32_16x16x32_bf16(false, av, false, bv3, (short)0, acc3, false, false);

        if (q + 1 < nq) waitAsync1();         // older fill (current) consumed next
        __syncthreads();                      // all waves done reading buf cur
        ++q;
      }
    }
  }

  const int nc = lane & 15;
#pragma unroll
  for (int f = 0; f < 4; ++f) {
    int nf = nBase + f * 16 + nc;
    if (nf >= Cout) continue;
    float bsv = bias[nf];
    v8f* accp = (f == 0) ? &acc0 : (f == 1) ? &acc1 : (f == 2) ? &acc2 : &acc3;
#pragma unroll
    for (int r = 0; r < 8; ++r) {
      int mm = tileM * 16 + ((lane < 16) ? r : r + 8);
      if (mm < M) out[(size_t)mm * Cout + nf] = (*accp)[r] + bsv;
    }
  }
}

// ---------------------------------------------------------------------------
// Small-Cin conv (first layer, Cin=3): direct loads, 16x32 tile.
// ---------------------------------------------------------------------------
__global__ __launch_bounds__(256) void conv_wmma_small(
    const __bf16* __restrict__ in, const __bf16* __restrict__ wrep,
    const float* __restrict__ bias, float* __restrict__ out,
    int B, int H, int W, int Cin, int Cout, int K, int stride,
    int Hout, int Wout, int padLo, int KtotPad)
{
  const int lane  = threadIdx.x & 31;
  const int wave  = threadIdx.x >> 5;
  const int tileM = blockIdx.x * 8 + wave;
  const int M     = B * Hout * Wout;

  const int mr = lane & 15;
  const int m  = tileM * 16 + mr;
  int bb = 0, oh = 0, ow = 0;
  if (m < M) { bb = m / (Hout * Wout); int r = m % (Hout * Wout); oh = r / Wout; ow = r % Wout; }
  const int aKoff = (lane < 16) ? 0 : 16;
  const int bKoff = (lane < 16) ? 0 : 16;
  const int n0 = blockIdx.y * 32 + (lane & 15);
  const int n1 = n0 + 16;
  const __bf16* wp0 = wrep + (size_t)(n0 < Cout ? n0 : 0) * KtotPad;
  const __bf16* wp1 = wrep + (size_t)(n1 < Cout ? n1 : 0) * KtotPad;

  v8f acc0 = {}, acc1 = {};
  for (int kh = 0; kh < K; ++kh) {
    int ih = oh * stride + kh - padLo;
    for (int kw = 0; kw < K; ++kw) {
      int iw = ow * stride + kw - padLo;
      bool ok = (m < M) && ((unsigned)ih < (unsigned)H) && ((unsigned)iw < (unsigned)W);
      const __bf16* ap = in + (((size_t)bb * H + ih) * W + iw) * Cin;
      int kbase = (kh * K + kw) * Cin;
      for (int c0 = 0; c0 < Cin; c0 += 32) {
        int kb = kbase + c0 + bKoff;
        __builtin_prefetch(wp0 + kb + 32, 0, 1);
        v16bf av, bv0, bv1;
#pragma unroll
        for (int j = 0; j < 16; ++j) {
          av[j] = (__bf16)0.0f;
          int ci = c0 + aKoff + j;
          if (ok && ci < Cin) av[j] = ap[ci];
          bv0[j] = wp0[kb + j];   // rows padded: in-bounds; A=0 masks tail
          bv1[j] = wp1[kb + j];
        }
        acc0 = __builtin_amdgcn_wmma_f32_16x16x32_bf16(false, av, false, bv0, (short)0, acc0, false, false);
        acc1 = __builtin_amdgcn_wmma_f32_16x16x32_bf16(false, av, false, bv1, (short)0, acc1, false, false);
      }
    }
  }

  float b0 = (n0 < Cout) ? bias[n0] : 0.f;
  float b1 = (n1 < Cout) ? bias[n1] : 0.f;
#pragma unroll
  for (int r = 0; r < 8; ++r) {
    int mm = tileM * 16 + ((lane < 16) ? r : r + 8);
    if (mm < M) {
      if (n0 < Cout) out[(size_t)mm * Cout + n0] = acc0[r] + b0;
      if (n1 < Cout) out[(size_t)mm * Cout + n1] = acc1[r] + b1;
    }
  }
}

// ---------------------------------------------------------------------------
// GroupNorm statistics: one block per (batch, group); 32 groups.
// ---------------------------------------------------------------------------
__global__ void gn_stats(const float* __restrict__ x, float* __restrict__ stats,
                         int H, int W, int C)
{
  int b = blockIdx.x / 32, g = blockIdx.x % 32;
  int Cpg = C / 32;
  long cnt = (long)H * W * Cpg;
  float s = 0.f, ss = 0.f;
  for (long i = threadIdx.x; i < cnt; i += blockDim.x) {
    long hw = i / Cpg; int c = g * Cpg + (int)(i % Cpg);
    float v = x[((long)b * H * W + hw) * C + c];
    s += v; ss += v * v;
  }
  __shared__ float rs[256], rss[256];
  rs[threadIdx.x] = s; rss[threadIdx.x] = ss; __syncthreads();
  for (int off = 128; off > 0; off >>= 1) {
    if ((int)threadIdx.x < off) {
      rs[threadIdx.x]  += rs[threadIdx.x + off];
      rss[threadIdx.x] += rss[threadIdx.x + off];
    }
    __syncthreads();
  }
  if (threadIdx.x == 0) {
    float mu  = rs[0] / (float)cnt;
    float var = rss[0] / (float)cnt - mu * mu;
    stats[(b * 32 + g) * 2 + 0] = mu;
    stats[(b * 32 + g) * 2 + 1] = var;
  }
}

// GN + LeakyReLU (+ optional bf16 residual) -> bf16 activation
__global__ void gn_apply(const float* __restrict__ raw, const float* __restrict__ gamma,
                         const float* __restrict__ beta, const float* __restrict__ stats,
                         const __bf16* __restrict__ res, __bf16* __restrict__ dst,
                         long HWC, int C, int Cpg, long N)
{
  long i = (long)blockIdx.x * blockDim.x + threadIdx.x;
  if (i >= N) return;
  int c = (int)(i % C);
  int b = (int)(i / HWC);
  int g = c / Cpg;
  float mu  = stats[(b * 32 + g) * 2 + 0];
  float var = stats[(b * 32 + g) * 2 + 1];
  float v = (raw[i] - mu) * rsqrtf(var + GN_EPS) * gamma[c] + beta[c];
  v = v > 0.f ? v : 0.1f * v;
  if (res) v += (float)res[i];
  dst[i] = (__bf16)v;
}

__global__ void leaky_inplace(float* x, long N)
{
  long i = (long)blockIdx.x * blockDim.x + threadIdx.x;
  if (i < N) { float v = x[i]; x[i] = v > 0.f ? v : 0.1f * v; }
}

// nearest-2x upsample of a (B,H,W,C1) concat with skip (B,2H,2W,C2)  [bf16]
__global__ void upsample_concat(const __bf16* __restrict__ a, const __bf16* __restrict__ skip,
                                __bf16* __restrict__ out, int B, int H, int W, int C1, int C2)
{
  int Cs = C1 + C2;
  long N = (long)B * 2 * H * 2 * W * Cs;
  long i = (long)blockIdx.x * blockDim.x + threadIdx.x;
  if (i >= N) return;
  int c = (int)(i % Cs); long t = i / Cs;
  int owi = (int)(t % (2 * W)); t /= (2 * W);
  int ohi = (int)(t % (2 * H)); int b = (int)(t / (2 * H));
  __bf16 v;
  if (c < C1) v = a[(((long)b * H + ohi / 2) * W + owi / 2) * C1 + c];
  else        v = skip[(((long)b * 2 * H + ohi) * 2 * W + owi) * C2 + (c - C1)];
  out[i] = v;
}

__device__ inline float dsig(float x) { return 1.f / (1.f + __expf(-x)); }

// Decode one scale: boxes + obj*softmax(class) scores.
__global__ void decode_kernel(const float* __restrict__ act, int B, int H, int W,
                              int candOff, float a0w, float a0h, float a1w, float a1h,
                              float a2w, float a2h,
                              float* __restrict__ bbox, float* __restrict__ scores)
{
  int total = B * H * W * 3;
  int i = blockIdx.x * blockDim.x + threadIdx.x;
  if (i >= total) return;
  int ai = i % 3; int t = i / 3;
  int ww = t % W; t /= W;
  int hh = t % H; int b = t / H;
  const float* p = act + ((((size_t)b * H + hh) * W + ww) * 3 + ai) * 85;
  float aw = (ai == 0 ? a0w : ai == 1 ? a1w : a2w) / 416.f;
  float ah = (ai == 0 ? a0h : ai == 1 ? a1h : a2h) / 416.f;
  float bx = dsig(p[0]) + (float)ww;
  float by = dsig(p[1]) + (float)hh;
  float bw = __expf(p[2]) * aw;
  float bh = __expf(p[3]) * ah;
  long cand = (long)b * 10647 + candOff + ((hh * W + ww) * 3 + ai);
  bbox[cand * 4 + 0] = (bx - 0.5f * bw) * 416.f;
  bbox[cand * 4 + 1] = (by - 0.5f * bh) * 416.f;
  bbox[cand * 4 + 2] = (bx + 0.5f * bw) * 416.f;
  bbox[cand * 4 + 3] = (by + 0.5f * bh) * 416.f;
  float conf = dsig(p[4]);
  float mx = -1e30f;
  for (int k = 0; k < 80; ++k) mx = fmaxf(mx, p[5 + k]);
  float sum = 0.f;
  for (int k = 0; k < 80; ++k) sum += __expf(p[5 + k] - mx);
  float inv = conf / sum;
  for (int k = 0; k < 80; ++k) scores[cand * 80 + k] = __expf(p[5 + k] - mx) * inv;
}

__device__ inline float iou_pair(const float* A, const float* Bx)
{
  float areaA = (A[2] - A[0]) * (A[3] - A[1]);
  float areaB = (Bx[2] - Bx[0]) * (Bx[3] - Bx[1]);
  float lx = fmaxf(A[0], Bx[0]), ly = fmaxf(A[1], Bx[1]);
  float rx = fminf(A[2], Bx[2]), ry = fminf(A[3], Bx[3]);
  float w = fmaxf(rx - lx, 0.f), h = fmaxf(ry - ly, 0.f);
  float inter = w * h;
  return inter / fmaxf(areaA + areaB - inter, 1e-9f);
}

// One block per (image, class): top-100 + greedy IoU suppression.
__global__ void nms_class(const float* __restrict__ scores, const float* __restrict__ bbox,
                          float* __restrict__ scr, float* __restrict__ keptB,
                          float* __restrict__ keptS)
{
  int blk = blockIdx.x; int b = blk / 80; int c = blk % 80;
  float* s0 = scr + (size_t)blk * 10647;
  for (int i = threadIdx.x; i < 10647; i += blockDim.x) {
    float v = scores[((size_t)b * 10647 + i) * 80 + c];
    s0[i] = (v > 0.5f) ? v : -1.0f;
  }
  __syncthreads();
  __shared__ float redV[256]; __shared__ int redI[256];
  __shared__ float selS[100]; __shared__ int selI[100];
  __shared__ float bbs[100][4]; __shared__ int keep[100];
  for (int t = 0; t < 100; ++t) {
    float bv = -1e30f; int bi = 0;
    for (int i = threadIdx.x; i < 10647; i += blockDim.x) {
      float v = s0[i];
      if (v > bv || (v == bv && i < bi)) { bv = v; bi = i; }
    }
    redV[threadIdx.x] = bv; redI[threadIdx.x] = bi; __syncthreads();
    for (int off = 128; off > 0; off >>= 1) {
      if ((int)threadIdx.x < off) {
        float v2 = redV[threadIdx.x + off]; int i2 = redI[threadIdx.x + off];
        if (v2 > redV[threadIdx.x] || (v2 == redV[threadIdx.x] && i2 < redI[threadIdx.x])) {
          redV[threadIdx.x] = v2; redI[threadIdx.x] = i2;
        }
      }
      __syncthreads();
    }
    if (threadIdx.x == 0) { selS[t] = redV[0]; selI[t] = redI[0]; s0[redI[0]] = -1e30f; }
    __syncthreads();
  }
  if (threadIdx.x < 100) {
    long gi = (long)b * 10647 + selI[threadIdx.x];
    for (int j = 0; j < 4; ++j) bbs[threadIdx.x][j] = bbox[gi * 4 + j];
    keep[threadIdx.x] = 1;
  }
  __syncthreads();
  for (int i = 0; i < 100; ++i) {
    bool alive = keep[i] && (selS[i] > 0.f);
    if (alive && (int)threadIdx.x > i && threadIdx.x < 100 && keep[threadIdx.x]) {
      if (iou_pair(bbs[i], bbs[threadIdx.x]) > 0.5f) keep[threadIdx.x] = 0;
    }
    __syncthreads();
  }
  if (threadIdx.x < 100) {
    int t = threadIdx.x;
    keptS[(size_t)blk * 100 + t] = (keep[t] && selS[t] > 0.f) ? selS[t] : -1.0f;
    for (int j = 0; j < 4; ++j) keptB[((size_t)blk * 100 + t) * 4 + j] = bbs[t][j];
  }
}

// One block per image: merge 80x100 -> final top-100.
__global__ void final_topk(const float* __restrict__ keptS, const float* __restrict__ keptB,
                           float* __restrict__ scrImg, float* __restrict__ outB,
                           float* __restrict__ outS, float* __restrict__ outC,
                           float* __restrict__ outV)
{
  int b = blockIdx.x;
  float* s = scrImg + (size_t)b * 8000;
  for (int i = threadIdx.x; i < 8000; i += blockDim.x) s[i] = keptS[(size_t)b * 8000 + i];
  __syncthreads();
  __shared__ float rv[256]; __shared__ int ri[256]; __shared__ int cnt;
  if (threadIdx.x == 0) cnt = 0;
  __syncthreads();
  for (int t = 0; t < 100; ++t) {
    float bv = -1e30f; int bi = 0;
    for (int i = threadIdx.x; i < 8000; i += blockDim.x) {
      float v = s[i];
      if (v > bv || (v == bv && i < bi)) { bv = v; bi = i; }
    }
    rv[threadIdx.x] = bv; ri[threadIdx.x] = bi; __syncthreads();
    for (int off = 128; off > 0; off >>= 1) {
      if ((int)threadIdx.x < off) {
        if (rv[threadIdx.x + off] > rv[threadIdx.x] ||
            (rv[threadIdx.x + off] == rv[threadIdx.x] && ri[threadIdx.x + off] < ri[threadIdx.x])) {
          rv[threadIdx.x] = rv[threadIdx.x + off]; ri[threadIdx.x] = ri[threadIdx.x + off];
        }
      }
      __syncthreads();
    }
    if (threadIdx.x == 0) {
      float v = rv[0]; int idx = ri[0];
      s[idx] = -1e30f;
      bool valid = v > 0.f;
      if (valid) cnt++;
      outS[b * 100 + t] = valid ? v : 0.f;
      outC[b * 100 + t] = valid ? (float)(idx / 100) : 0.f;
      for (int j = 0; j < 4; ++j)
        outB[(b * 100 + t) * 4 + j] = valid ? keptB[((size_t)b * 8000 + idx) * 4 + j] : 0.f;
    }
    __syncthreads();
  }
  if (threadIdx.x == 0) outV[b] = (float)cnt;
}

// ---------------------------------------------------------------------------
// Host driver
// ---------------------------------------------------------------------------
struct ConvP { const float *w, *b, *g, *be; };

extern "C" void kernel_launch(void* const* d_in, const int* in_sizes, int n_in,
                              void* d_out, int out_size, void* d_ws, size_t ws_size,
                              hipStream_t stream)
{
  (void)in_sizes; (void)n_in; (void)out_size; (void)ws_size;
  const float* xin = (const float*)d_in[0];
  const int B = 2;

  // ---- walk params in jax pytree (sorted dict key) order ----
  int cur = 3;
  auto nextGN = [&]() { ConvP p; p.b = (const float*)d_in[cur++]; p.be = (const float*)d_in[cur++];
                        p.g = (const float*)d_in[cur++]; p.w = (const float*)d_in[cur++]; return p; };
  auto nextPl = [&]() { ConvP p; p.b = (const float*)d_in[cur++]; p.g = nullptr; p.be = nullptr;
                        p.w = (const float*)d_in[cur++]; return p; };
  int nres[5] = {1, 2, 8, 8, 4};
  ConvP down[5], resP[23][2]; int ri = 0;
  for (int blk = 0; blk < 5; ++blk) {
    down[blk] = nextGN();
    for (int r = 0; r < nres[blk]; ++r) { resP[ri][0] = nextGN(); resP[ri][1] = nextGN(); ++ri; }
  }
  ConvP dn1 = nextGN();
  ConvP yc1[5]; for (int i = 0; i < 5; ++i) yc1[i] = nextGN();
  ConvP yc2[6]; for (int i = 0; i < 6; ++i) yc2[i] = nextGN();
  ConvP yc3[6]; for (int i = 0; i < 6; ++i) yc3[i] = nextGN();
  ConvP yo1c1 = nextGN(); ConvP yo1c2 = nextPl();
  ConvP yo2c1 = nextGN(); ConvP yo2c2 = nextPl();
  ConvP yo3c1 = nextGN(); ConvP yo3c2 = nextPl();

  // ---- workspace layout (byte allocator, 256B aligned chunks) ----
  char* wsBase = (char*)d_ws; size_t off = 0;
  auto allocB = [&](size_t bytes) { void* p = wsBase + off; off += (bytes + 255) & ~(size_t)255; return p; };
  float*  rawB = (float*)allocB(11075584ull * 4);      // f32 conv output
  __bf16* actX = (__bf16*)allocB(11075584ull * 2);
  __bf16* actT = (__bf16*)allocB(11075584ull * 2);
  __bf16* actU = (__bf16*)allocB(5537792ull * 2);
  __bf16* inBf = (__bf16*)allocB(1038336ull * 2);
  __bf16* x36  = (__bf16*)allocB(1384448ull * 2);
  __bf16* x61  = (__bf16*)allocB(692224ull * 2);
  __bf16* t1   = (__bf16*)allocB(173056ull * 2);
  __bf16* t2   = (__bf16*)allocB(346112ull * 2);
  __bf16* t3   = (__bf16*)allocB(692224ull * 2);
  __bf16* wrep = (__bf16*)allocB(4800000ull * 2);      // max 9*512*1024 + pad
  float* bboxBuf   = (float*)allocB((size_t)B * 10647 * 4 * 4);
  float* scoresBuf = (float*)allocB((size_t)B * 10647 * 80 * 4);
  float* scrCls    = (float*)allocB((size_t)160 * 10647 * 4);
  float* keptB     = (float*)allocB(64000ull * 4);
  float* keptS     = (float*)allocB(16000ull * 4);
  float* scrImg    = (float*)allocB(16000ull * 4);
  float* stats     = (float*)allocB(128ull * 4);

  float* OUT = (float*)d_out;
  float* o0 = OUT;                  // (2,13,13,255)
  float* o1 = OUT + 86190;          // (2,26,26,255)
  float* o2 = OUT + 430950;         // (2,52,52,255)
  float* oB = OUT + 1809990;
  float* oS = oB + 800;
  float* oC = oS + 200;
  float* oV = oC + 200;

  auto conv = [&](const __bf16* src, const ConvP& p, float* raw,
                  int H, int W, int Cin, int Cout, int K, int stride) {
    int Ktot = K * K * Cin;
    int KtotPad = ((Ktot + 63) / 32) * 32;            // +extra 32 for tail b128 reads
    long Nw = (long)Cout * KtotPad;
    repack_w<<<dim3((unsigned)((Nw + 255) / 256)), 256, 0, stream>>>(p.w, wrep, Ktot, KtotPad, Cout);
    int Ho = (stride == 1) ? H : H / 2;
    int Wo = (stride == 1) ? W : W / 2;
    int padTot = (Ho - 1) * stride + K - H; if (padTot < 0) padTot = 0;
    int padLo = padTot / 2;
    int M = B * Ho * Wo;
    if ((Cin & 31) == 0) {
      dim3 g((((M + 15) / 16) + 7) / 8, (Cout + 63) / 64);
      conv_wmma_fast<<<g, 256, 0, stream>>>(src, wrep, p.b, raw, B, H, W, Cin, Cout,
                                            K, stride, Ho, Wo, padLo, KtotPad);
    } else {
      dim3 g((((M + 15) / 16) + 7) / 8, (Cout + 31) / 32);
      conv_wmma_small<<<g, 256, 0, stream>>>(src, wrep, p.b, raw, B, H, W, Cin, Cout,
                                             K, stride, Ho, Wo, padLo, KtotPad);
    }
  };
  auto gn = [&](float* raw, const ConvP& p, __bf16* dst, const __bf16* res,
                int H, int W, int C) {
    gn_stats<<<dim3(B * 32), 256, 0, stream>>>(raw, stats, H, W, C);
    long N = (long)B * H * W * C;
    gn_apply<<<dim3((unsigned)((N + 255) / 256)), 256, 0, stream>>>(
        raw, p.g, p.be, stats, res, dst, (long)H * W * C, C, C / 32, N);
  };
  auto dconvF = [&](const __bf16* src, const ConvP& p, __bf16* dst,
                    int H, int W, int Cin, int Cout, int K, int stride,
                    const __bf16* res) {
    conv(src, p, rawB, H, W, Cin, Cout, K, stride);
    int Ho = (stride == 1) ? H : H / 2;
    int Wo = (stride == 1) ? W : W / 2;
    gn(rawB, p, dst, res, Ho, Wo, Cout);
  };

  // ---- input f32 -> bf16 ----
  f32_to_bf16<<<dim3((1038336 + 255) / 256), 256, 0, stream>>>(xin, inBf, 1038336);

  // ---- backbone ----
  dconvF(inBf, dn1, actX, 416, 416, 3, 32, 3, 1, nullptr);
  int Hc = 416, Wc = 416, Cc = 32;
  int filters[5] = {64, 128, 256, 512, 1024};
  ri = 0;
  for (int blk = 0; blk < 5; ++blk) {
    int f = filters[blk];
    dconvF(actX, down[blk], actX, Hc, Wc, Cc, f, 3, 2, nullptr);
    Hc /= 2; Wc /= 2; Cc = f;
    for (int r = 0; r < nres[blk]; ++r) {
      dconvF(actX, resP[ri][0], actT, Hc, Wc, f, f / 2, 1, 1, nullptr);
      dconvF(actT, resP[ri][1], actX, Hc, Wc, f / 2, f, 3, 1, actX);
      ++ri;
    }
    if (blk == 2) (void)hipMemcpyAsync(x36, actX, (size_t)B * 52 * 52 * 256 * 2, hipMemcpyDeviceToDevice, stream);
    if (blk == 3) (void)hipMemcpyAsync(x61, actX, (size_t)B * 26 * 26 * 512 * 2, hipMemcpyDeviceToDevice, stream);
  }

  // ---- yc1 / yo1 (13x13) ----
  dconvF(actX, yc1[0], actT, 13, 13, 1024, 512, 1, 1, nullptr);
  dconvF(actT, yc1[1], actU, 13, 13, 512, 1024, 3, 1, nullptr);
  dconvF(actU, yc1[2], actT, 13, 13, 1024, 512, 1, 1, nullptr);
  dconvF(actT, yc1[3], actU, 13, 13, 512, 1024, 3, 1, nullptr);
  dconvF(actU, yc1[4], t1, 13, 13, 1024, 512, 1, 1, nullptr);
  dconvF(t1, yo1c1, actT, 13, 13, 512, 1024, 3, 1, nullptr);
  conv(actT, yo1c2, o0, 13, 13, 1024, 255, 1, 1);
  leaky_inplace<<<dim3((86190 + 255) / 256), 256, 0, stream>>>(o0, 86190);

  // ---- yc2 / yo2 (26x26) ----
  dconvF(t1, yc2[0], actT, 13, 13, 512, 256, 1, 1, nullptr);
  {
    long N = (long)B * 26 * 26 * 768;
    upsample_concat<<<dim3((unsigned)((N + 255) / 256)), 256, 0, stream>>>(
        actT, x61, actU, B, 13, 13, 256, 512);
  }
  dconvF(actU, yc2[1], actT, 26, 26, 768, 256, 1, 1, nullptr);
  dconvF(actT, yc2[2], actU, 26, 26, 256, 512, 3, 1, nullptr);
  dconvF(actU, yc2[3], actT, 26, 26, 512, 256, 1, 1, nullptr);
  dconvF(actT, yc2[4], actU, 26, 26, 256, 512, 3, 1, nullptr);
  dconvF(actU, yc2[5], t2, 26, 26, 512, 256, 1, 1, nullptr);
  dconvF(t2, yo2c1, actT, 26, 26, 256, 512, 3, 1, nullptr);
  conv(actT, yo2c2, o1, 26, 26, 512, 255, 1, 1);
  leaky_inplace<<<dim3((344760 + 255) / 256), 256, 0, stream>>>(o1, 344760);

  // ---- yc3 / yo3 (52x52) ----
  dconvF(t2, yc3[0], actT, 26, 26, 256, 128, 1, 1, nullptr);
  {
    long N = (long)B * 52 * 52 * 384;
    upsample_concat<<<dim3((unsigned)((N + 255) / 256)), 256, 0, stream>>>(
        actT, x36, actU, B, 26, 26, 128, 256);
  }
  dconvF(actU, yc3[1], actT, 52, 52, 384, 128, 1, 1, nullptr);
  dconvF(actT, yc3[2], actU, 52, 52, 128, 256, 3, 1, nullptr);
  dconvF(actU, yc3[3], actT, 52, 52, 256, 128, 1, 1, nullptr);
  dconvF(actT, yc3[4], actU, 52, 52, 128, 256, 3, 1, nullptr);
  dconvF(actU, yc3[5], t3, 52, 52, 256, 128, 1, 1, nullptr);
  dconvF(t3, yo3c1, actT, 52, 52, 128, 256, 3, 1, nullptr);
  conv(actT, yo3c2, o2, 52, 52, 256, 255, 1, 1);
  leaky_inplace<<<dim3((1379040 + 255) / 256), 256, 0, stream>>>(o2, 1379040);

  // ---- decode (bbox + scores) ----
  decode_kernel<<<dim3((B * 13 * 13 * 3 + 255) / 256), 256, 0, stream>>>(
      o0, B, 13, 13, 0, 116.f, 90.f, 156.f, 198.f, 373.f, 326.f, bboxBuf, scoresBuf);
  decode_kernel<<<dim3((B * 26 * 26 * 3 + 255) / 256), 256, 0, stream>>>(
      o1, B, 26, 26, 507, 30.f, 61.f, 62.f, 45.f, 59.f, 119.f, bboxBuf, scoresBuf);
  decode_kernel<<<dim3((B * 52 * 52 * 3 + 255) / 256), 256, 0, stream>>>(
      o2, B, 52, 52, 2535, 10.f, 13.f, 16.f, 30.f, 33.f, 23.f, bboxBuf, scoresBuf);

  // ---- NMS ----
  nms_class<<<dim3(B * 80), 256, 0, stream>>>(scoresBuf, bboxBuf, scrCls, keptB, keptS);
  final_topk<<<dim3(B), 256, 0, stream>>>(keptS, keptB, scrImg, oB, oS, oC, oV);
}
